// EnhancedTransformerBlock_62551903699103
// MI455X (gfx1250) — 16-bit and 8-bit matrix operand order repaired and checked by reading, compiled, not run on hardware
//
#include <hip/hip_runtime.h>
#include <hip/hip_bf16.h>
#include <stdint.h>

#define S    1024
#define DM   4096
#define NH   64
#define HD   64
#define MLPD 16384
#define INV_SF (1.0f / 45.254834f)

typedef __attribute__((ext_vector_type(8)))  int       v8i;
typedef __attribute__((ext_vector_type(8)))  float     v8f;
typedef __attribute__((ext_vector_type(16))) _Float16  v16h;

__device__ __forceinline__ float red_max16(float v) {
#pragma unroll
  for (int m = 1; m < 16; m <<= 1) v = fmaxf(v, __shfl_xor(v, m, 16));
  return v;
}
__device__ __forceinline__ float red_sum16(float v) {
#pragma unroll
  for (int m = 1; m < 16; m <<= 1) v += __shfl_xor(v, m, 16);
  return v;
}
__device__ __forceinline__ float gelu_tanh(float x) {
  float c = x + 0.044715f * x * x * x;
  return 0.5f * x * (1.0f + tanhf(0.7978845608f * c));
}
__device__ __forceinline__ int8_t quant1(float x, float s) {
  float v = fminf(fmaxf(x / s, -127.0f), 127.0f);
  return (int8_t)rintf(v);
}

__global__ void k_zero16(unsigned* amax) {
  if (threadIdx.x < 16) amax[threadIdx.x] = 0u;
}
__global__ void k_fin(const unsigned* amax, float* scl, int lo, int hi) {
  int i = lo + (int)threadIdx.x;
  if (i < hi) scl[i] = __uint_as_float(amax[i]) / 127.0f + 1e-8f;
}
__global__ __launch_bounds__(256) void k_absmax(const float* __restrict__ x,
                                                size_t n, unsigned* amax) {
  __shared__ float red[256];
  float lm = 0.0f;
  for (size_t i = (size_t)blockIdx.x * 256 + threadIdx.x; i < n;
       i += (size_t)gridDim.x * 256)
    lm = fmaxf(lm, fabsf(x[i]));
  red[threadIdx.x] = lm;
  __syncthreads();
  for (int s = 128; s > 0; s >>= 1) {
    if ((int)threadIdx.x < s) red[threadIdx.x] = fmaxf(red[threadIdx.x], red[threadIdx.x + s]);
    __syncthreads();
  }
  if (threadIdx.x == 0) atomicMax(amax, __float_as_uint(red[0]));
}

__global__ __launch_bounds__(256) void k_rmsnorm(const float* __restrict__ x,
                                                 const float* __restrict__ g,
                                                 float* __restrict__ y,
                                                 unsigned* amax) {
  __shared__ float red[256];
  const size_t row = blockIdx.x;
  const float* xr = x + row * DM;
  float* yr = y + row * DM;
  float ss = 0.0f;
  for (int c = threadIdx.x; c < DM; c += 256) { float v = xr[c]; ss += v * v; }
  red[threadIdx.x] = ss;
  __syncthreads();
  for (int s = 128; s > 0; s >>= 1) {
    if ((int)threadIdx.x < s) red[threadIdx.x] += red[threadIdx.x + s];
    __syncthreads();
  }
  const float rinv = rsqrtf(red[0] / (float)DM + 1e-6f);
  __syncthreads();
  float lm = 0.0f;
  for (int c = threadIdx.x; c < DM; c += 256) {
    float v = xr[c] * rinv * g[c];
    yr[c] = v;
    lm = fmaxf(lm, fabsf(v));
  }
  red[threadIdx.x] = lm;
  __syncthreads();
  for (int s = 128; s > 0; s >>= 1) {
    if ((int)threadIdx.x < s) red[threadIdx.x] = fmaxf(red[threadIdx.x], red[threadIdx.x + s]);
    __syncthreads();
  }
  if (threadIdx.x == 0) atomicMax(amax, __float_as_uint(red[0]));
}

__global__ __launch_bounds__(256) void k_quant(const float* __restrict__ x,
                                               const float* __restrict__ scl,
                                               int8_t* __restrict__ q, size_t n) {
  const float s = scl[0];
  for (size_t i = (size_t)blockIdx.x * 256 + threadIdx.x; i < n;
       i += (size_t)gridDim.x * 256)
    q[i] = quant1(x[i], s);
}

__global__ __launch_bounds__(256) void k_quant_heads(
    const float* __restrict__ qf, const float* __restrict__ kf,
    const float* __restrict__ vf, const float* __restrict__ sq,
    const float* __restrict__ sk, const float* __restrict__ sv,
    int8_t* __restrict__ q8, int8_t* __restrict__ k8,
    _Float16* __restrict__ v16) {
  const float a = sq[0], b = sk[0], c = sv[0];
  const size_t n = (size_t)S * DM;
  for (size_t i = (size_t)blockIdx.x * 256 + threadIdx.x; i < n;
       i += (size_t)gridDim.x * 256) {
    size_t s = i >> 12;
    int col = (int)(i & 4095);
    int h = col >> 6, d = col & 63;
    size_t dst = ((size_t)h * S + s) * 64 + d;
    q8[dst] = quant1(qf[i], a);
    k8[dst] = quant1(kf[i], b);
    v16[dst] = (_Float16)rintf(fminf(fmaxf(vf[i] / c, -127.0f), 127.0f));
  }
}

__global__ __launch_bounds__(256) void k_qtrans(const float* __restrict__ W,
                                                const float* __restrict__ scl,
                                                int8_t* __restrict__ Wt,
                                                int K, int N) {
  __shared__ float tile[32][33];
  const int n0 = blockIdx.x * 32, k0 = blockIdx.y * 32;
  const int tx = threadIdx.x & 31, ty = threadIdx.x >> 5;
#pragma unroll
  for (int t = 0; t < 4; ++t)
    tile[ty + t * 8][tx] = W[(size_t)(k0 + ty + t * 8) * N + n0 + tx];
  __syncthreads();
  const float s = scl[0];
#pragma unroll
  for (int t = 0; t < 4; ++t) {
    int n = n0 + ty + t * 8, k = k0 + tx;
    Wt[(size_t)n * K + k] = quant1(tile[tx][ty + t * 8], s);
  }
}

__global__ __launch_bounds__(256) void k_gemm_i8(
    const int8_t* __restrict__ A, const int8_t* __restrict__ Bt,
    const float* __restrict__ bias, const float* __restrict__ sA,
    const float* __restrict__ sB, const float* __restrict__ resid,
    float* __restrict__ C, int M, int N, int K, int do_gelu) {
  const int lane = threadIdx.x & 31;
  const int wid = threadIdx.x >> 5;
  const int nl = lane & 15, hi = lane >> 4;
  const int row0 = blockIdx.y * 256 + (wid & 3) * 64;
  const int col0 = blockIdx.x * 64 + (wid >> 2) * 32;

  v8i acc[4][2];
#pragma unroll
  for (int i = 0; i < 4; ++i)
#pragma unroll
    for (int j = 0; j < 2; ++j)
#pragma unroll
      for (int r = 0; r < 8; ++r) acc[i][j][r] = 0;

  for (int k = 0; k < K; k += 64) {
    v8i a[4];
#pragma unroll
    for (int mt = 0; mt < 4; ++mt) {
      const int8_t* ap = A + (size_t)(row0 + mt * 16 + nl) * K + k + hi * 8;
      __builtin_prefetch(ap + 64, 0, 3);
#pragma unroll
      for (int v = 0; v < 8; ++v) {
        const int off = ((v >> 1) << 4) + ((v & 1) << 2);
        a[mt][v] = *(const int*)(ap + off);
      }
    }
    v8i b[2];
#pragma unroll
    for (int nt = 0; nt < 2; ++nt) {
      const int8_t* bp = Bt + (size_t)(col0 + nt * 16 + nl) * K + k + hi * 8;
      __builtin_prefetch(bp + 64, 0, 3);
#pragma unroll
      for (int v = 0; v < 8; ++v) {
        const int off = ((v >> 1) << 4) + ((v & 1) << 2);
        b[nt][v] = *(const int*)(bp + off);
      }
    }
#pragma unroll
    for (int mt = 0; mt < 4; ++mt)
#pragma unroll
      for (int nt = 0; nt < 2; ++nt)
        acc[mt][nt] = __builtin_amdgcn_wmma_i32_16x16x64_iu8(
            true, a[mt], true, b[nt], acc[mt][nt], false, false);
  }

  const float ss = sA[0] * sB[0];
#pragma unroll
  for (int mt = 0; mt < 4; ++mt)
#pragma unroll
    for (int nt = 0; nt < 2; ++nt) {
      const int col = col0 + nt * 16 + nl;
      const float bv = bias[col];
#pragma unroll
      for (int r = 0; r < 8; ++r) {
        const int row = row0 + mt * 16 + r + hi * 8;
        float f = (float)acc[mt][nt][r] * ss + bv;
        if (resid) f += resid[(size_t)row * N + col];
        if (do_gelu) f = gelu_tanh(f);
        C[(size_t)row * N + col] = f;
      }
    }
}

__global__ __launch_bounds__(32) void k_attn(
    const int8_t* __restrict__ qh, const int8_t* __restrict__ kh,
    const _Float16* __restrict__ vh, const float* __restrict__ relw,
    const float* __restrict__ sq, const float* __restrict__ sk,
    const float* __restrict__ sv, float* __restrict__ ao) {
  __shared__ _Float16 pl[16 * 32];
  const int lane = threadIdx.x;
  const int nl = lane & 15, hi = lane >> 4;
  const int h = blockIdx.y;
  const int q0 = blockIdx.x * 16;

  v8i aq;
  {
    const int8_t* qp = qh + ((size_t)h * S + q0 + nl) * 64 + hi * 8;
#pragma unroll
    for (int v = 0; v < 8; ++v)
      aq[v] = *(const int*)(qp + ((v >> 1) << 4) + ((v & 1) << 2));
  }
  const float sqk = sq[0] * sk[0];

  float rm[8], rs[8];
  v8f o[4];
#pragma unroll
  for (int r = 0; r < 8; ++r) { rm[r] = -3.0e38f; rs[r] = 0.0f; }
#pragma unroll
  for (int j = 0; j < 4; ++j)
#pragma unroll
    for (int r = 0; r < 8; ++r) o[j][r] = 0.0f;

  for (int kc = 0; kc < S; kc += 32) {
    v8i zc;
#pragma unroll
    for (int r = 0; r < 8; ++r) zc[r] = 0;
    v8i b0, b1;
    {
      const int8_t* kp0 = kh + ((size_t)h * S + kc + nl) * 64 + hi * 8;
      const int8_t* kp1 = kp0 + 16 * 64;
#pragma unroll
      for (int v = 0; v < 8; ++v) {
        const int off = ((v >> 1) << 4) + ((v & 1) << 2);
        b0[v] = *(const int*)(kp0 + off);
        b1[v] = *(const int*)(kp1 + off);
      }
    }
    v8i s0 = __builtin_amdgcn_wmma_i32_16x16x64_iu8(true, aq, true, b0, zc, false, false);
    v8i s1 = __builtin_amdgcn_wmma_i32_16x16x64_iu8(true, aq, true, b1, zc, false, false);

    float p0[8], p1[8];
#pragma unroll
    for (int r = 0; r < 8; ++r) {
      const int qrow = q0 + r + hi * 8;
      const int k0i = kc + nl, k1i = kc + 16 + nl;
      const int pos0 = min(max(qrow - k0i + 32, 0), 64);
      const int pos1 = min(max(qrow - k1i + 32, 0), 64);
      float f0 = ((float)s0[r] * sqk + relw[pos0 * 64 + h]) * INV_SF;
      float f1 = ((float)s1[r] * sqk + relw[pos1 * 64 + h]) * INV_SF;
      float tm = red_max16(fmaxf(f0, f1));
      float nm = fmaxf(rm[r], tm);
      float corr = __expf(rm[r] - nm);
      float e0 = __expf(f0 - nm), e1 = __expf(f1 - nm);
      rs[r] = rs[r] * corr + red_sum16(e0 + e1);
      rm[r] = nm;
#pragma unroll
      for (int j = 0; j < 4; ++j) o[j][r] *= corr;
      p0[r] = e0;
      p1[r] = e1;
    }

    __syncthreads();
#pragma unroll
    for (int r = 0; r < 8; ++r) {
      pl[(r + hi * 8) * 32 + nl] = (_Float16)p0[r];
      pl[(r + hi * 8) * 32 + 16 + nl] = (_Float16)p1[r];
    }
    __syncthreads();
    union { v16h v; int i[8]; } pa;
    {
      const int* pli = (const int*)pl;
#pragma unroll
      for (int j = 0; j < 8; ++j) {
        const int base = ((j >> 2) << 4) + hi * 8 + ((j & 3) << 1);
        pa.i[j] = pli[(nl * 32 + base) >> 1];
      }
    }

#pragma unroll
    for (int j = 0; j < 4; ++j) {
      union { v16h v; _Float16 e[16]; } vb;
      const _Float16* vpb =
          vh + ((size_t)h * S + kc + hi * 8) * 64 + j * 16 + nl;
#pragma unroll
      for (int t = 0; t < 8; ++t) {
        vb.e[2 * t] = vpb[(2 * t + ((t >> 2) << 3)) * 64];
        vb.e[2 * t + 1] = vpb[(2 * t + 1 + ((t >> 2) << 3)) * 64];
      }
      o[j] = __builtin_amdgcn_wmma_f32_16x16x32_f16(false, pa.v, false, vb.v,
                                                    (short)0, o[j], false, false);
    }
    __syncthreads();
  }

  const float svv = sv[0];
#pragma unroll
  for (int r = 0; r < 8; ++r) {
    const float fac = svv / (rs[r] + 1e-6f);
    const int row = q0 + r + hi * 8;
#pragma unroll
    for (int j = 0; j < 4; ++j)
      ao[(size_t)row * DM + h * 64 + j * 16 + nl] = o[j][r] * fac;
  }
}

extern "C" void kernel_launch(void* const* d_in, const int* in_sizes, int n_in,
                              void* d_out, int out_size, void* d_ws,
                              size_t ws_size, hipStream_t stream) {
  const float* inp = (const float*)d_in[0];
  const float* g1  = (const float*)d_in[1];
  const float* wq  = (const float*)d_in[2];
  const float* bq  = (const float*)d_in[3];
  const float* wk  = (const float*)d_in[4];
  const float* bk  = (const float*)d_in[5];
  const float* wv  = (const float*)d_in[6];
  const float* bv  = (const float*)d_in[7];
  const float* rel = (const float*)d_in[8];
  const float* wo  = (const float*)d_in[9];
  const float* bo  = (const float*)d_in[10];
  const float* g2  = (const float*)d_in[11];
  const float* w1  = (const float*)d_in[12];
  const float* b1  = (const float*)d_in[13];
  const float* w2  = (const float*)d_in[14];
  const float* b2  = (const float*)d_in[15];
  float* out = (float*)d_out;

  char* ws = (char*)d_ws;
  size_t off = 0;
  auto alloc = [&](size_t bytes) -> char* {
    char* p = ws + off;
    off = (off + bytes + 255) & ~(size_t)255;
    return p;
  };
  unsigned* amax = (unsigned*)alloc(16 * sizeof(unsigned));
  float*    scl  = (float*)alloc(16 * sizeof(float));
  float*   xn1  = (float*)alloc((size_t)S * DM * 4);
  int8_t*  x8   = (int8_t*)alloc((size_t)S * DM);
  int8_t*  wqT  = (int8_t*)alloc((size_t)DM * DM);
  int8_t*  wkT  = (int8_t*)alloc((size_t)DM * DM);
  int8_t*  wvT  = (int8_t*)alloc((size_t)DM * DM);
  int8_t*  woT  = (int8_t*)alloc((size_t)DM * DM);
  int8_t*  w1T  = (int8_t*)alloc((size_t)MLPD * DM);
  int8_t*  w2T  = (int8_t*)alloc((size_t)DM * MLPD);
  float*   qf   = (float*)alloc((size_t)S * DM * 4);
  float*   kf   = (float*)alloc((size_t)S * DM * 4);
  float*   vf   = (float*)alloc((size_t)S * DM * 4);
  int8_t*  q8   = (int8_t*)alloc((size_t)S * DM);
  int8_t*  k8   = (int8_t*)alloc((size_t)S * DM);
  _Float16* v16p = (_Float16*)alloc((size_t)S * DM * 2);
  float*   aob  = (float*)alloc((size_t)S * DM * 4);
  int8_t*  ao8  = (int8_t*)alloc((size_t)S * DM);
  float*   x1   = (float*)alloc((size_t)S * DM * 4);
  float*   xn2  = (float*)alloc((size_t)S * DM * 4);
  int8_t*  xn28 = (int8_t*)alloc((size_t)S * DM);
  float*   hb   = (float*)alloc((size_t)S * MLPD * 4);
  int8_t*  h8   = (int8_t*)alloc((size_t)S * MLPD);
  (void)ws_size; (void)in_sizes; (void)n_in; (void)out_size;

  k_zero16<<<1, 32, 0, stream>>>(amax);
  k_rmsnorm<<<S, 256, 0, stream>>>(inp, g1, xn1, amax + 0);
  k_absmax<<<512, 256, 0, stream>>>(wq, (size_t)DM * DM, amax + 1);
  k_absmax<<<512, 256, 0, stream>>>(wk, (size_t)DM * DM, amax + 2);
  k_absmax<<<512, 256, 0, stream>>>(wv, (size_t)DM * DM, amax + 3);
  k_absmax<<<512, 256, 0, stream>>>(wo, (size_t)DM * DM, amax + 4);
  k_absmax<<<1024, 256, 0, stream>>>(w1, (size_t)DM * MLPD, amax + 5);
  k_absmax<<<1024, 256, 0, stream>>>(w2, (size_t)MLPD * DM, amax + 6);
  k_fin<<<1, 16, 0, stream>>>(amax, scl, 0, 7);

  k_quant<<<2048, 256, 0, stream>>>(xn1, scl + 0, x8, (size_t)S * DM);
  k_qtrans<<<dim3(DM / 32, DM / 32), 256, 0, stream>>>(wq, scl + 1, wqT, DM, DM);
  k_qtrans<<<dim3(DM / 32, DM / 32), 256, 0, stream>>>(wk, scl + 2, wkT, DM, DM);
  k_qtrans<<<dim3(DM / 32, DM / 32), 256, 0, stream>>>(wv, scl + 3, wvT, DM, DM);
  k_qtrans<<<dim3(DM / 32, DM / 32), 256, 0, stream>>>(wo, scl + 4, woT, DM, DM);
  k_qtrans<<<dim3(MLPD / 32, DM / 32), 256, 0, stream>>>(w1, scl + 5, w1T, DM, MLPD);
  k_qtrans<<<dim3(DM / 32, MLPD / 32), 256, 0, stream>>>(w2, scl + 6, w2T, MLPD, DM);

  dim3 gq(DM / 64, S / 256);
  k_gemm_i8<<<gq, 256, 0, stream>>>(x8, wqT, bq, scl + 0, scl + 1, nullptr, qf, S, DM, DM, 0);
  k_gemm_i8<<<gq, 256, 0, stream>>>(x8, wkT, bk, scl + 0, scl + 2, nullptr, kf, S, DM, DM, 0);
  k_gemm_i8<<<gq, 256, 0, stream>>>(x8, wvT, bv, scl + 0, scl + 3, nullptr, vf, S, DM, DM, 0);

  k_absmax<<<512, 256, 0, stream>>>(qf, (size_t)S * DM, amax + 7);
  k_absmax<<<512, 256, 0, stream>>>(kf, (size_t)S * DM, amax + 8);
  k_absmax<<<512, 256, 0, stream>>>(vf, (size_t)S * DM, amax + 9);
  k_fin<<<1, 16, 0, stream>>>(amax, scl, 7, 10);
  k_quant_heads<<<2048, 256, 0, stream>>>(qf, kf, vf, scl + 7, scl + 8, scl + 9,
                                          q8, k8, v16p);

  k_attn<<<dim3(S / 16, NH), 32, 0, stream>>>(q8, k8, v16p, rel, scl + 7,
                                              scl + 8, scl + 9, aob);

  k_absmax<<<512, 256, 0, stream>>>(aob, (size_t)S * DM, amax + 10);
  k_fin<<<1, 16, 0, stream>>>(amax, scl, 10, 11);
  k_quant<<<2048, 256, 0, stream>>>(aob, scl + 10, ao8, (size_t)S * DM);
  k_gemm_i8<<<gq, 256, 0, stream>>>(ao8, woT, bo, scl + 10, scl + 4, inp, x1, S, DM, DM, 0);

  k_rmsnorm<<<S, 256, 0, stream>>>(x1, g2, xn2, amax + 11);
  k_fin<<<1, 16, 0, stream>>>(amax, scl, 11, 12);
  k_quant<<<2048, 256, 0, stream>>>(xn2, scl + 11, xn28, (size_t)S * DM);
  dim3 gm(MLPD / 64, S / 256);
  k_gemm_i8<<<gm, 256, 0, stream>>>(xn28, w1T, b1, scl + 11, scl + 5, nullptr, hb, S, MLPD, DM, 1);
  k_absmax<<<2048, 256, 0, stream>>>(hb, (size_t)S * MLPD, amax + 12);
  k_fin<<<1, 16, 0, stream>>>(amax, scl, 12, 13);
  k_quant<<<4096, 256, 0, stream>>>(hb, scl + 12, h8, (size_t)S * MLPD);
  k_gemm_i8<<<gq, 256, 0, stream>>>(h8, w2T, b2, scl + 12, scl + 6, x1, out, S, DM, MLPD, 0);
}
